// PrototypeAutoencoder_81630148428094
// MI455X (gfx1250) — matrix operand order repaired and checked by reading, not compiled, not run on hardware
//
#include <hip/hip_runtime.h>
#include <hip/hip_bf16.h>
#include <math.h>

typedef __attribute__((ext_vector_type(16))) _Float16 v16h;
typedef __attribute__((ext_vector_type(8)))  float    v8f;
typedef __attribute__((ext_vector_type(2)))  float    v2f;

// ---------------------------------------------------------------------------
// STN conv1 (3->16ch, 3x3 SAME) + relu + maxpool2 : [16,3,32,32] -> [16,16,16,16]
// ---------------------------------------------------------------------------
__global__ __launch_bounds__(256) void stn1_kernel(const float* __restrict__ protos,
        const float* __restrict__ w, const float* __restrict__ bias,
        float* __restrict__ f1) {
  __shared__ float img[3][32][32];
  __shared__ float wl[27];
  int p = blockIdx.x >> 4, co = blockIdx.x & 15;
  int tid = threadIdx.x;
  for (int i = tid; i < 3 * 32 * 32; i += 256) ((float*)img)[i] = protos[p * 3072 + i];
  if (tid < 27) wl[tid] = w[co * 27 + tid];
  __syncthreads();
  float bb = bias[co];
  int py = tid >> 4, px = tid & 15;
  float best = -1e30f;
  for (int dy = 0; dy < 2; ++dy)
    for (int dx = 0; dx < 2; ++dx) {
      int y = py * 2 + dy, x0 = px * 2 + dx;
      float s = bb;
      for (int ci = 0; ci < 3; ++ci)
        for (int ty = 0; ty < 3; ++ty)
          for (int tx = 0; tx < 3; ++tx) {
            int yy = y + ty - 1, xx = x0 + tx - 1;
            if (yy >= 0 && yy < 32 && xx >= 0 && xx < 32)
              s += img[ci][yy][xx] * wl[ci * 9 + ty * 3 + tx];
          }
      best = fmaxf(best, s);
    }
  f1[((p * 16 + co) * 16 + py) * 16 + px] = fmaxf(best, 0.f);
}

// ---------------------------------------------------------------------------
// STN conv2 (16->32ch) + relu + maxpool2 : [16,16,16,16] -> [16,32,8,8]
// ---------------------------------------------------------------------------
__global__ __launch_bounds__(64) void stn2_kernel(const float* __restrict__ f1,
        const float* __restrict__ w, const float* __restrict__ bias,
        float* __restrict__ f2) {
  __shared__ float img[16][16][16];
  __shared__ float wl[144];
  int p = blockIdx.x >> 5, co = blockIdx.x & 31;
  int tid = threadIdx.x;
  for (int i = tid; i < 4096; i += 64) ((float*)img)[i] = f1[p * 4096 + i];
  for (int i = tid; i < 144; i += 64) wl[i] = w[co * 144 + i];
  __syncthreads();
  float bb = bias[co];
  int py = tid >> 3, px = tid & 7;
  float best = -1e30f;
  for (int dy = 0; dy < 2; ++dy)
    for (int dx = 0; dx < 2; ++dx) {
      int y = py * 2 + dy, x0 = px * 2 + dx;
      float s = bb;
      for (int ci = 0; ci < 16; ++ci)
        for (int ty = 0; ty < 3; ++ty)
          for (int tx = 0; tx < 3; ++tx) {
            int yy = y + ty - 1, xx = x0 + tx - 1;
            if (yy >= 0 && yy < 16 && xx >= 0 && xx < 16)
              s += img[ci][yy][xx] * wl[ci * 9 + ty * 3 + tx];
          }
      best = fmaxf(best, s);
    }
  f2[((p * 32 + co) * 8 + py) * 8 + px] = fmaxf(best, 0.f);
}

// ---------------------------------------------------------------------------
// STN fc (2048->32, relu) + per-proto loc head -> theta[16][6]
// ---------------------------------------------------------------------------
__global__ __launch_bounds__(64) void stnfc_kernel(const float* __restrict__ f2,
        const float* __restrict__ fcw, const float* __restrict__ fcb,
        const float* __restrict__ locw, const float* __restrict__ locb,
        float* __restrict__ theta) {
  __shared__ float fo[32];
  int p = blockIdx.x, tid = threadIdx.x;
  if (tid < 32) {
    float s = fcb[tid];
    const float* fp = f2 + p * 2048;
    const float* wr = fcw + tid * 2048;
    for (int i = 0; i < 2048; ++i) s += fp[i] * wr[i];
    fo[tid] = fmaxf(s, 0.f);
  }
  __syncthreads();
  if (tid < 6) {
    float s = locb[p * 6 + tid];
    for (int j = 0; j < 32; ++j) s += locw[(p * 6 + tid) * 32 + j] * fo[j];
    theta[p * 6 + tid] = s;
  }
}

// ---------------------------------------------------------------------------
// affine grid + bilinear grid_sample (zeros pad, align_corners) -> tp[16,3,32,32]
// ---------------------------------------------------------------------------
__global__ __launch_bounds__(256) void gridsample_kernel(const float* __restrict__ protos,
        const float* __restrict__ theta, float* __restrict__ tp) {
  int p = blockIdx.x, tid = threadIdx.x;
  __shared__ float th[6];
  if (tid < 6) th[tid] = theta[p * 6 + tid];
  __syncthreads();
  const float* img = protos + p * 3072;
  for (int q = 0; q < 4; ++q) {
    int pix = tid + q * 256;
    int hh = pix >> 5, ww = pix & 31;
    float xs = -1.f + 2.f * (float)ww / 31.f;
    float ys = -1.f + 2.f * (float)hh / 31.f;
    float gx = th[0] * xs + th[1] * ys + th[2];
    float gy = th[3] * xs + th[4] * ys + th[5];
    float ix = (gx + 1.f) * 0.5f * 31.f;
    float iy = (gy + 1.f) * 0.5f * 31.f;
    float x0 = floorf(ix), y0 = floorf(iy);
    float x1 = x0 + 1.f, y1 = y0 + 1.f;
    float wgt[4] = { (x1 - ix) * (y1 - iy), (x1 - ix) * (iy - y0),
                     (ix - x0) * (y1 - iy), (ix - x0) * (iy - y0) };
    float xf[4] = { x0, x0, x1, x1 };
    float yf[4] = { y0, y1, y0, y1 };
    float acc[3] = { 0.f, 0.f, 0.f };
    for (int c4 = 0; c4 < 4; ++c4) {
      bool valid = xf[c4] >= 0.f && xf[c4] <= 31.f && yf[c4] >= 0.f && yf[c4] <= 31.f;
      float wv = valid ? wgt[c4] : 0.f;
      int xc = (int)fminf(fmaxf(xf[c4], 0.f), 31.f);
      int yc = (int)fminf(fmaxf(yf[c4], 0.f), 31.f);
      for (int c = 0; c < 3; ++c) acc[c] += img[c * 1024 + yc * 32 + xc] * wv;
    }
    for (int c = 0; c < 3; ++c) tp[p * 3072 + c * 1024 + pix] = acc[c];
  }
}

// ---------------------------------------------------------------------------
// Fused encoder: conv1(WMMA f16) -> relu -> conv2(WMMA f16) -> relu -> mean ->
// fc -> sigmoid -> clip.  One block (8 waves) per image; h1 never leaves LDS.
//
// Implicit GEMM, im2col k = ci*9 + tap.  Pad k-values are routed to dedicated
// ZEROED LDS channels (xp[3], hp[16..17]) so the A-fragment gather has NO
// lane-divergent predicate -> unconditional ds_load_u16 stream, no exec-mask
// branching, one dscnt wait per fragment.
// ---------------------------------------------------------------------------
__global__ __launch_bounds__(256) void encoder_kernel(
    const float* __restrict__ x,
    const float* __restrict__ w1, const float* __restrict__ b1,
    const float* __restrict__ w2, const float* __restrict__ b2,
    const float* __restrict__ fcw, const float* __restrict__ fcb,
    float* __restrict__ wts, float* __restrict__ out_w) {
  __shared__ _Float16 xp[4][34][34];    // padded input, channel 3 = zeros (k-pad)
  __shared__ _Float16 Bc1[32][16];      // conv1 B matrix [K=32pad][N=16]
  __shared__ _Float16 hp[18][34][34];   // padded conv1 out, ch 16..17 = zeros (k-pad)
  __shared__ _Float16 B2[160][32];      // conv2 B matrix [K=160pad][N=32]
  __shared__ float sums[32];            // per-channel spatial sums
  int b = blockIdx.x, tid = threadIdx.x;

  __builtin_prefetch(x + b * 3072, 0, 1);            // global_prefetch_b8

  // zero hp entirely (covers SAME-pad border + zero channels 16..17)
  for (int i = tid; i < (18 * 34 * 34) / 2; i += 256) ((unsigned int*)hp)[i] = 0u;
  // xp: channels 0..2 = padded image, channel 3 = zeros
  for (int i = tid; i < 4 * 34 * 34; i += 256) {
    int c = i / 1156, r = i % 1156;
    int y = r / 34 - 1, xx = r % 34 - 1;
    float v = (c < 3 && y >= 0 && y < 32 && xx >= 0 && xx < 32)
                  ? x[(b * 3 + c) * 1024 + y * 32 + xx] : 0.f;
    ((_Float16*)xp)[i] = (_Float16)v;
  }
  for (int i = tid; i < 512; i += 256) {             // conv1 weights, k = ci*9+tap
    int k = i >> 4, n = i & 15;
    Bc1[k][n] = (k < 27) ? (_Float16)w1[n * 27 + k] : (_Float16)0.f;
  }
  for (int i = tid; i < 5120; i += 256) {            // conv2 weights, k = ci*9+tap
    int k = i >> 5, n = i & 31;
    B2[k][n] = (k < 144) ? (_Float16)w2[n * 144 + k] : (_Float16)0.f;
  }
  if (tid < 32) sums[tid] = 0.f;
  __syncthreads();

  int lane = tid & 31, wave = tid >> 5;
  int mlo = lane & 15;
  int hi = (lane >= 16) ? 1 : 0;
  int kbase = hi ? 8 : 0;

  // ---- conv1: one 16x16x32 WMMA per 16-pixel tile, N=16 out channels ----
  v16h bf1;
  #pragma unroll
  for (int i = 0; i < 16; ++i) bf1[i] = Bc1[kbase + (i < 8 ? i : i + 8)][mlo];
  float bias1 = b1[mlo];
  for (int t = 0; t < 8; ++t) {
    int tile = wave * 8 + t;
    int m = tile * 16 + mlo, py = m >> 5, px = m & 31;
    v16h a;
    #pragma unroll
    for (int i = 0; i < 16; ++i) {
      int k = kbase + (i < 8 ? i : i + 8);           // ISA 16-bit A layout
      int ci = k / 9; int tap = k - ci * 9;          // pad k -> ci==3 (zero channel)
      int ty = tap / 3, tx = tap - ty * 3;
      a[i] = xp[ci][py + ty][px + tx];
    }
    v8f c = {};
    c = __builtin_amdgcn_wmma_f32_16x16x32_f16(false, a, false, bf1, (short)0, c, false, false);
    int moff = tile * 16 + (hi ? 8 : 0);
    #pragma unroll
    for (int r = 0; r < 8; ++r) {
      int mm = moff + r;
      float v = fmaxf(c[r] + bias1, 0.f);
      hp[mlo][1 + (mm >> 5)][1 + (mm & 31)] = (_Float16)v;
    }
  }
  __syncthreads();

  // ---- conv2: 5 k-steps x 2 n-tiles of 16x16x32 WMMA per pixel tile ----
  v16h bf[5][2];
  #pragma unroll
  for (int kf = 0; kf < 5; ++kf)
    #pragma unroll
    for (int nt = 0; nt < 2; ++nt)
      #pragma unroll
      for (int i = 0; i < 16; ++i)
        bf[kf][nt][i] = B2[kf * 32 + kbase + (i < 8 ? i : i + 8)][nt * 16 + mlo];
  float bias2a = b2[mlo], bias2b = b2[16 + mlo];
  float acc0 = 0.f, acc1 = 0.f;
  for (int t = 0; t < 8; ++t) {
    int tile = wave * 8 + t;
    int m = tile * 16 + mlo, py = m >> 5, px = m & 31;
    v8f c0 = {}, c1 = {};
    #pragma unroll
    for (int kf = 0; kf < 5; ++kf) {
      v16h a;
      #pragma unroll
      for (int i = 0; i < 16; ++i) {
        int k = kf * 32 + kbase + (i < 8 ? i : i + 8);
        int ci = k / 9; int tap = k - ci * 9;        // pad k -> ci in {16,17} (zeros)
        int ty = tap / 3, tx = tap - ty * 3;
        a[i] = hp[ci][py + ty][px + tx];
      }
      c0 = __builtin_amdgcn_wmma_f32_16x16x32_f16(false, a, false, bf[kf][0], (short)0, c0, false, false);
      c1 = __builtin_amdgcn_wmma_f32_16x16x32_f16(false, a, false, bf[kf][1], (short)0, c1, false, false);
    }
    #pragma unroll
    for (int r = 0; r < 8; ++r) {
      acc0 += fmaxf(c0[r] + bias2a, 0.f);
      acc1 += fmaxf(c1[r] + bias2b, 0.f);
    }
  }
  atomicAdd(&sums[mlo], acc0);
  atomicAdd(&sums[16 + mlo], acc1);
  __syncthreads();

  // ---- mean + fc + sigmoid + clip ----
  if (tid < 16) {
    float v = fcb[tid];
    for (int c = 0; c < 32; ++c) v += (sums[c] * (1.f / 1024.f)) * fcw[tid * 32 + c];
    float s = 1.f / (1.f + expf(-v));
    s = fminf(fmaxf(s, 1e-4f), 1.f - 1e-4f);
    wts[b * 16 + tid] = s;
    out_w[b * 16 + tid] = s;
  }
}

// ---------------------------------------------------------------------------
// recon = weights[1024,16] @ tp[16,3072] via V_WMMA_F32_16X16X4_F32 (4 k-steps)
// 1536 blocks x 8 waves = 12288 tiles (64 M-tiles x 192 N-tiles)
// ---------------------------------------------------------------------------
__global__ __launch_bounds__(256) void recon_kernel(const float* __restrict__ wts,
        const float* __restrict__ tp, float* __restrict__ out) {
  int tid = threadIdx.x, lane = tid & 31, wave = tid >> 5;
  int t = blockIdx.x * 8 + wave;
  int mtile = t / 192, ntile = t % 192;
  int mlo = lane & 15;
  int hi = (lane >= 16) ? 1 : 0;
  int koff = hi ? 2 : 0;
  v8f c = {};
  #pragma unroll
  for (int ks = 0; ks < 4; ++ks) {
    v2f a, bv;
    #pragma unroll
    for (int j = 0; j < 2; ++j) {
      int k = ks * 4 + koff + j;                     // ISA 32-bit A 16x4 / B 4x16 layout
      a[j] = wts[(mtile * 16 + mlo) * 16 + k];
      bv[j] = tp[k * 3072 + ntile * 16 + mlo];
    }
    c = __builtin_amdgcn_wmma_f32_16x16x4_f32(false, a, false, bv, (short)0, c, false, false);
  }
  int n = ntile * 16 + mlo;
  #pragma unroll
  for (int r = 0; r < 8; ++r) {
    int mm = mtile * 16 + r + (hi ? 8 : 0);
    out[mm * 3072 + n] = c[r];
  }
}

// ---------------------------------------------------------------------------
// broadcast tp[16,3,32,32] -> out[1024,16,3,32,32] (float4 streaming stores)
// ---------------------------------------------------------------------------
__global__ __launch_bounds__(256) void bc_protos_kernel(const float4* __restrict__ tp4,
        float4* __restrict__ out4) {
  int j = blockIdx.x * 256 + (int)threadIdx.x;
  out4[(size_t)blockIdx.y * 12288 + j] = tp4[j];
}

__global__ __launch_bounds__(96) void bc_thetas_kernel(const float* __restrict__ theta,
        float* __restrict__ out) {
  out[(size_t)blockIdx.x * 96 + threadIdx.x] = theta[threadIdx.x];
}

// ---------------------------------------------------------------------------
extern "C" void kernel_launch(void* const* d_in, const int* in_sizes, int n_in,
                              void* d_out, int out_size, void* d_ws, size_t ws_size,
                              hipStream_t stream) {
  const float* x      = (const float*)d_in[0];
  const float* protos = (const float*)d_in[1];
  const float* e1w = (const float*)d_in[2];
  const float* e1b = (const float*)d_in[3];
  const float* e2w = (const float*)d_in[4];
  const float* e2b = (const float*)d_in[5];
  const float* efw = (const float*)d_in[6];
  const float* efb = (const float*)d_in[7];
  const float* s1w = (const float*)d_in[8];
  const float* s1b = (const float*)d_in[9];
  const float* s2w = (const float*)d_in[10];
  const float* s2b = (const float*)d_in[11];
  const float* sfw = (const float*)d_in[12];
  const float* sfb = (const float*)d_in[13];
  const float* lw  = (const float*)d_in[14];
  const float* lb  = (const float*)d_in[15];

  float* out_recon = (float*)d_out;                 // 1024*3*32*32  = 3145728
  float* out_w     = out_recon + 3145728;           // 1024*16       = 16384
  float* out_tp    = out_w + 16384;                 // 1024*16*3072  = 50331648
  float* out_th    = out_tp + 50331648;             // 1024*16*6     = 98304

  float* ws        = (float*)d_ws;
  float* wts_ws    = ws;                            // 16384
  float* tp_ws     = ws + 16384;                    // 49152 (64KB offset, 16B aligned)
  float* theta_ws  = ws + 65536;                    // 96
  float* f1_ws     = ws + 65664;                    // 16*16*16*16 = 65536
  float* f2_ws     = ws + 131200;                   // 16*32*8*8   = 16384

  // STN path (tiny, VALU)
  stn1_kernel<<<256, 256, 0, stream>>>(protos, s1w, s1b, f1_ws);
  stn2_kernel<<<512, 64, 0, stream>>>(f1_ws, s2w, s2b, f2_ws);
  stnfc_kernel<<<16, 64, 0, stream>>>(f2_ws, sfw, sfb, lw, lb, theta_ws);
  gridsample_kernel<<<16, 256, 0, stream>>>(protos, theta_ws, tp_ws);

  // Encoder (fused conv1+conv2+pool+fc, WMMA f16)
  encoder_kernel<<<1024, 256, 0, stream>>>(x, e1w, e1b, e2w, e2b, efw, efb,
                                           wts_ws, out_w);

  // Weighted reconstruction (WMMA f32 16x16x4)
  recon_kernel<<<1536, 256, 0, stream>>>(wts_ws, tp_ws, out_recon);

  // Broadcast outputs (pure bandwidth)
  bc_protos_kernel<<<dim3(48, 1024), 256, 0, stream>>>((const float4*)tp_ws,
                                                       (float4*)out_tp);
  bc_thetas_kernel<<<1024, 96, 0, stream>>>(theta_ws, out_th);
}
